// CNN_42228118454858
// MI455X (gfx1250) — compile-verified, matrix operand order checked by reading, not run on hardware
//
#include <hip/hip_runtime.h>
#include <math.h>

typedef __attribute__((ext_vector_type(16))) _Float16 v16h;
typedef __attribute__((ext_vector_type(8)))  _Float16 h8;
typedef __attribute__((ext_vector_type(8)))  float    v8f;
typedef __attribute__((ext_vector_type(4)))  unsigned u32x4;
typedef __attribute__((ext_vector_type(8)))  int      i32x8;
typedef __attribute__((ext_vector_type(4)))  int      i32x4;

#define B_    64
#define L_    512
#define E_    256
#define P_    64
#define KOUT  512
#define KS_   3
#define H_    512
#define T_    53
#define CIN   896      // 3*E + 2*P
#define RTOT  2688     // CIN*KS
#define LP    514      // L + 2  (zero row on each side)
#define COM   2048     // 6*E + H

// A-tile staged per block: 32 output rows + 2 overlap rows of padded sent
#define AROWS 34
#define AHALF (AROWS * CIN)        // 30464 halves = 60928 bytes
#define AQW   (AHALF / 4)          // 7616 8-byte elements

#if defined(__has_builtin)
#if __has_builtin(__builtin_amdgcn_tensor_load_to_lds) && \
    __has_builtin(__builtin_amdgcn_s_wait_tensorcnt)
#define HAVE_TDM 1
#endif
#endif

union V16 { v16h v; h8 h[2]; };

// ---- order-preserving float <-> uint map (deterministic atomic max) ----
__device__ __forceinline__ unsigned enc_f32(float f) {
  unsigned u = __float_as_uint(f);
  return (u & 0x80000000u) ? ~u : (u | 0x80000000u);
}
__device__ __forceinline__ float dec_f32(unsigned u) {
  return (u & 0x80000000u) ? __uint_as_float(u ^ 0x80000000u)
                           : __uint_as_float(~u);
}

// ---- 1) repack conv_w (K, Cin, KS) f32 -> w2[k][dk*Cin + c] f16 ----
__global__ void repack_w_kernel(const float* __restrict__ conv_w,
                                _Float16* __restrict__ w2) {
  int i = blockIdx.x * blockDim.x + threadIdx.x;
  if (i >= KOUT * RTOT) return;
  int k = i / RTOT;
  int r = i % RTOT;
  int dk = r / CIN;
  int c  = r % CIN;
  w2[i] = (_Float16)conv_w[((size_t)k * CIN + c) * KS_ + dk];
}

// ---- 2) build padded sent (B, LP, CIN) in f16 ----
__global__ void build_sent_kernel(const int* __restrict__ context,
                                  const int* __restrict__ subject_dis,
                                  const int* __restrict__ object_dis,
                                  const float* __restrict__ embed_table,
                                  const float* __restrict__ pos_table,
                                  _Float16* __restrict__ sp) {
  size_t i = (size_t)blockIdx.x * blockDim.x + threadIdx.x;
  const size_t total = (size_t)B_ * LP * CIN;
  if (i >= total) return;
  int c  = (int)(i % CIN);
  size_t t = i / CIN;
  int lp = (int)(t % LP);
  int b  = (int)(t / LP);
  float f = 0.0f;
  if (lp >= 1 && lp <= L_) {
    int l = lp - 1;
    if (c < 3 * E_) {
      int seg = c / E_;
      int e   = c % E_;
      int ls  = l - 1 + seg;
      if (ls >= 0 && ls < L_)
        f = embed_table[(size_t)context[b * L_ + ls] * E_ + e];
    } else if (c < 3 * E_ + P_) {
      f = pos_table[(size_t)subject_dis[b * L_ + l] * P_ + (c - 3 * E_)];
    } else {
      f = pos_table[(size_t)object_dis[b * L_ + l] * P_ + (c - 3 * E_ - P_)];
    }
  }
  sp[i] = (_Float16)f;
}

// ---- 3) init encoded pooled maxima ----
__global__ void init_pooled_kernel(unsigned* __restrict__ pooled_enc) {
  int i = blockIdx.x * blockDim.x + threadIdx.x;
  if (i < B_ * KOUT) pooled_enc[i] = 0u;  // decodes below any finite value
}

// ---- 4) conv-as-GEMM via WMMA + TDM-staged A tile, fused max-pool ----
// Block: (b, ltg) covering 32 l positions; 8 waves = 8 channel groups of 64.
// Wave tile: M=32 x N=64, reduction R=2688 in chunks of 32.
__global__ __launch_bounds__(256) void conv_wmma_kernel(
    const _Float16* __restrict__ sp, const _Float16* __restrict__ w2,
    unsigned* __restrict__ pooled_enc) {
  __shared__ _Float16 ldsA[AHALF];

  const int kg   = threadIdx.x >> 5;   // wave id = channel group
  const int lane = threadIdx.x & 31;
  const int hi   = lane >> 4;
  const int lm   = lane & 15;
  const int ltg  = blockIdx.x & 15;    // 16 groups of 32 l positions
  const int b    = blockIdx.x >> 4;    // batch

  const _Float16* gsrc = sp + ((size_t)b * LP + (size_t)ltg * 32) * CIN;

  // ---- stage the shared A tile (34 rows x 896) into LDS ----
#ifdef HAVE_TDM
  if (threadIdx.x < 32) {
    unsigned lds_off = (unsigned)(size_t)(&ldsA[0]);
    unsigned long long ga = (unsigned long long)(const void*)gsrc;
    u32x4 g0;
    g0[0] = 1u;                                    // count=1, user-mode load
    g0[1] = lds_off;                               // lds_addr
    g0[2] = (unsigned)(ga & 0xFFFFFFFFull);        // global_addr[31:0]
    g0[3] = (unsigned)((ga >> 32) & 0x01FFFFFFull) // global_addr[56:32]
            | (2u << 30);                          // type = 2 ("image")
    i32x8 g1;
    g1[0] = (int)(3u << 16);                       // data_size = 8 bytes
    g1[1] = (int)((AQW & 0xFFFFu) << 16);          // tensor_dim0[15:0]
    g1[2] = (int)(AQW >> 16);                      // tensor_dim0[31:16]
    g1[3] = (int)((AQW & 0xFFFFu) << 16);          // tile_dim0 = AQW (1-D)
    g1[4] = 0;                                     // tile_dim1/2 unused
    g1[5] = (int)AQW;                              // tensor_dim0_stride lo
    g1[6] = 0;
    g1[7] = 0;
    i32x4 g2 = {};
    i32x4 g3 = {};
    i32x8 g4 = {};
    __builtin_amdgcn_tensor_load_to_lds(g0, g1, g2, g3, g4, 0);
    __builtin_amdgcn_s_wait_tensorcnt(0);
  }
#else
  // cooperative fallback copy, 8 halves (16B) per thread per step
  {
    const h8* s = (const h8*)gsrc;
    h8* d = (h8*)ldsA;
    for (int i = threadIdx.x; i < AHALF / 8; i += 256) d[i] = s[i];
  }
#endif
  __syncthreads();

  // B: column n = channel kg*64 + t*16 + lm; rows contiguous along r
  const _Float16* bbase = w2 + (size_t)(kg * 64 + lm) * RTOT + hi * 8;
  // A rows in LDS: output row j (0..31) reads window starting at j*CIN
  const _Float16* a0 = ldsA + (size_t)(0 * 16 + lm) * CIN + hi * 8;
  const _Float16* a1 = ldsA + (size_t)(1 * 16 + lm) * CIN + hi * 8;

  v8f acc00 = {}, acc01 = {}, acc02 = {}, acc03 = {};
  v8f acc10 = {}, acc11 = {}, acc12 = {}, acc13 = {};

  for (int r0 = 0; r0 < RTOT; r0 += 32) {
    V16 A0, A1, b0, b1, b2, b3;
    A0.h[0] = *(const h8*)(a0 + r0);
    A0.h[1] = *(const h8*)(a0 + r0 + 16);
    A1.h[0] = *(const h8*)(a1 + r0);
    A1.h[1] = *(const h8*)(a1 + r0 + 16);
    b0.h[0] = *(const h8*)(bbase + r0);
    b0.h[1] = *(const h8*)(bbase + r0 + 16);
    b1.h[0] = *(const h8*)(bbase + (size_t)16 * RTOT + r0);
    b1.h[1] = *(const h8*)(bbase + (size_t)16 * RTOT + r0 + 16);
    b2.h[0] = *(const h8*)(bbase + (size_t)32 * RTOT + r0);
    b2.h[1] = *(const h8*)(bbase + (size_t)32 * RTOT + r0 + 16);
    b3.h[0] = *(const h8*)(bbase + (size_t)48 * RTOT + r0);
    b3.h[1] = *(const h8*)(bbase + (size_t)48 * RTOT + r0 + 16);

    acc00 = __builtin_amdgcn_wmma_f32_16x16x32_f16(false, A0.v, false, b0.v,
                                                   (short)0, acc00, false, false);
    acc01 = __builtin_amdgcn_wmma_f32_16x16x32_f16(false, A0.v, false, b1.v,
                                                   (short)0, acc01, false, false);
    acc02 = __builtin_amdgcn_wmma_f32_16x16x32_f16(false, A0.v, false, b2.v,
                                                   (short)0, acc02, false, false);
    acc03 = __builtin_amdgcn_wmma_f32_16x16x32_f16(false, A0.v, false, b3.v,
                                                   (short)0, acc03, false, false);
    acc10 = __builtin_amdgcn_wmma_f32_16x16x32_f16(false, A1.v, false, b0.v,
                                                   (short)0, acc10, false, false);
    acc11 = __builtin_amdgcn_wmma_f32_16x16x32_f16(false, A1.v, false, b1.v,
                                                   (short)0, acc11, false, false);
    acc12 = __builtin_amdgcn_wmma_f32_16x16x32_f16(false, A1.v, false, b2.v,
                                                   (short)0, acc12, false, false);
    acc13 = __builtin_amdgcn_wmma_f32_16x16x32_f16(false, A1.v, false, b3.v,
                                                   (short)0, acc13, false, false);
  }

  // max over M (=l): 8 accumulator rows x 2 sub-tiles, then fold lane^16
  float m0 = fmaxf(acc00[0], acc10[0]);
  float m1 = fmaxf(acc01[0], acc11[0]);
  float m2 = fmaxf(acc02[0], acc12[0]);
  float m3 = fmaxf(acc03[0], acc13[0]);
#pragma unroll
  for (int r = 1; r < 8; ++r) {
    m0 = fmaxf(m0, fmaxf(acc00[r], acc10[r]));
    m1 = fmaxf(m1, fmaxf(acc01[r], acc11[r]));
    m2 = fmaxf(m2, fmaxf(acc02[r], acc12[r]));
    m3 = fmaxf(m3, fmaxf(acc03[r], acc13[r]));
  }
  m0 = fmaxf(m0, __shfl_xor(m0, 16));
  m1 = fmaxf(m1, __shfl_xor(m1, 16));
  m2 = fmaxf(m2, __shfl_xor(m2, 16));
  m3 = fmaxf(m3, __shfl_xor(m3, 16));

  if (lane < 16) {
    unsigned* p = pooled_enc + (size_t)b * KOUT + kg * 64 + lm;
    atomicMax(p + 0,  enc_f32(m0));
    atomicMax(p + 16, enc_f32(m1));
    atomicMax(p + 32, enc_f32(m2));
    atomicMax(p + 48, enc_f32(m3));
  }
}

// ---- 5) entity features: [mean(span), left, right] -> (B, 3E) ----
__global__ void entity_kernel(const int* __restrict__ context,
                              const int* __restrict__ subject_idx,
                              const int* __restrict__ object_idx,
                              const float* __restrict__ embed_table,
                              float* __restrict__ subj,
                              float* __restrict__ obj) {
  int blk   = blockIdx.x;          // b*2 + which
  int which = blk & 1;
  int b     = blk >> 1;
  int e     = threadIdx.x;         // 0..255
  const int* idx = which ? object_idx : subject_idx;
  float* out     = which ? obj : subj;
  int s  = idx[b * 2 + 0];
  int en = idx[b * 2 + 1];
  float sum = 0.0f;
  for (int l = s; l <= en; ++l)
    sum += embed_table[(size_t)context[b * L_ + l] * E_ + e];
  float mean  = sum / (float)(en - s + 1);
  float left  = embed_table[(size_t)context[b * L_ + (s - 1)] * E_ + e];
  float right = (en + 1 < L_)
      ? embed_table[(size_t)context[b * L_ + (en + 1)] * E_ + e] : 0.0f;
  float* o = out + (size_t)b * (3 * E_);
  o[e]          = mean;
  o[E_ + e]     = left;
  o[2 * E_ + e] = right;
}

// ---- 6) sent_h = tanh((pooled+conv_b) @ lin1_w^T + lin1_b) ----
__global__ void lin1_kernel(const unsigned* __restrict__ pooled_enc,
                            const float* __restrict__ conv_b,
                            const float* __restrict__ lin1_w,
                            const float* __restrict__ lin1_b,
                            float* __restrict__ sent_h) {
  int i = blockIdx.x * blockDim.x + threadIdx.x;
  if (i >= B_ * H_) return;
  int b = i / H_, h = i % H_;
  const float*    wr = lin1_w + (size_t)h * KOUT;
  const unsigned* pr = pooled_enc + (size_t)b * KOUT;
  float acc = lin1_b[h];
  for (int k = 0; k < KOUT; ++k)
    acc += wr[k] * (dec_f32(pr[k]) + conv_b[k]);
  sent_h[i] = tanhf(acc);
}

// ---- 7) scores = [subj, obj, sent_h] @ lin2_w^T + lin2_b ----
__global__ void lin2_kernel(const float* __restrict__ subj,
                            const float* __restrict__ obj,
                            const float* __restrict__ sent_h,
                            const float* __restrict__ lin2_w,
                            const float* __restrict__ lin2_b,
                            float* __restrict__ out) {
  int i = blockIdx.x * blockDim.x + threadIdx.x;
  if (i >= B_ * T_) return;
  int b = i / T_, t = i % T_;
  const float* w = lin2_w + (size_t)t * COM;
  float acc = lin2_b[t];
  const float* sj = subj + (size_t)b * (3 * E_);
  const float* ob = obj  + (size_t)b * (3 * E_);
  const float* sh = sent_h + (size_t)b * H_;
  for (int j = 0; j < 3 * E_; ++j) acc += w[j] * sj[j];
  for (int j = 0; j < 3 * E_; ++j) acc += w[3 * E_ + j] * ob[j];
  for (int j = 0; j < H_; ++j)     acc += w[6 * E_ + j] * sh[j];
  out[(size_t)b * T_ + t] = acc;
}

extern "C" void kernel_launch(void* const* d_in, const int* in_sizes, int n_in,
                              void* d_out, int out_size, void* d_ws, size_t ws_size,
                              hipStream_t stream) {
  (void)in_sizes; (void)n_in; (void)out_size; (void)ws_size;
  const int*   context     = (const int*)d_in[0];
  const int*   subject_idx = (const int*)d_in[1];
  const int*   object_idx  = (const int*)d_in[2];
  const int*   subject_dis = (const int*)d_in[3];
  const int*   object_dis  = (const int*)d_in[4];
  const float* embed_table = (const float*)d_in[5];
  const float* pos_table   = (const float*)d_in[6];
  const float* conv_w      = (const float*)d_in[7];
  const float* conv_b      = (const float*)d_in[8];
  const float* lin1_w      = (const float*)d_in[9];
  const float* lin1_b      = (const float*)d_in[10];
  const float* lin2_w      = (const float*)d_in[11];
  const float* lin2_b      = (const float*)d_in[12];
  float* out = (float*)d_out;

  // workspace carve-out (~60 MB total)
  char* ws = (char*)d_ws;
  size_t off = 0;
  auto carve = [&](size_t bytes) -> void* {
    void* p = ws + off;
    off = (off + bytes + 255) & ~(size_t)255;
    return p;
  };
  _Float16* sp         = (_Float16*)carve(sizeof(_Float16) * (size_t)B_ * LP * CIN);
  _Float16* w2         = (_Float16*)carve(sizeof(_Float16) * (size_t)KOUT * RTOT);
  unsigned* pooled_enc = (unsigned*)carve(sizeof(unsigned) * (size_t)B_ * KOUT);
  float*    sent_h     = (float*)   carve(sizeof(float) * (size_t)B_ * H_);
  float*    subj       = (float*)   carve(sizeof(float) * (size_t)B_ * 3 * E_);
  float*    obj        = (float*)   carve(sizeof(float) * (size_t)B_ * 3 * E_);

  // 1) repack weights to f16
  {
    int n = KOUT * RTOT;
    repack_w_kernel<<<(n + 255) / 256, 256, 0, stream>>>(conv_w, w2);
  }
  // 2) build padded f16 sent tensor
  {
    size_t n = (size_t)B_ * LP * CIN;
    build_sent_kernel<<<(unsigned)((n + 255) / 256), 256, 0, stream>>>(
        context, subject_dis, object_dis, embed_table, pos_table, sp);
  }
  // 3) init pooled maxima
  {
    int n = B_ * KOUT;
    init_pooled_kernel<<<(n + 255) / 256, 256, 0, stream>>>(pooled_enc);
  }
  // 4) the 90-GFLOP conv GEMM on the WMMA pipe
  //    1024 blocks x 8 waves; A tile staged via Tensor Data Mover
  conv_wmma_kernel<<<B_ * 16, 256, 0, stream>>>(sp, w2, pooled_enc);
  // 5) entity span features
  entity_kernel<<<B_ * 2, E_, 0, stream>>>(context, subject_idx, object_idx,
                                           embed_table, subj, obj);
  // 6) lin1 + tanh
  {
    int n = B_ * H_;
    lin1_kernel<<<(n + 255) / 256, 256, 0, stream>>>(pooled_enc, conv_b,
                                                     lin1_w, lin1_b, sent_h);
  }
  // 7) final classifier
  {
    int n = B_ * T_;
    lin2_kernel<<<(n + 255) / 256, 256, 0, stream>>>(subj, obj, sent_h,
                                                     lin2_w, lin2_b, out);
  }
}
